// Interaction_59992103191102
// MI455X (gfx1250) — matrix operand order repaired and checked by reading, not compiled, not run on hardware
//
#include <hip/hip_runtime.h>
#include <hip/hip_bf16.h>

#define H 128
#define NB 8
#define NRAD 6
#define NSNR 42
#define PKELEMS (8*4*32*16)   // packed B matrix: [ntile=8][kchunk=4][lane=32][16 bf16]
#define LSTR 136              // LDS row stride (bf16 elems), 272B rows, 16B aligned

typedef __attribute__((ext_vector_type(16))) __bf16 v16bf;
typedef __attribute__((ext_vector_type(8)))  float  v8f;

__device__ __forceinline__ float silu(float v) {
  return v / (1.0f + __expf(-v));
}

union ABf16 { __bf16 b[16]; uint4 q[2]; v16bf v; };

__device__ __forceinline__ v8f wmma_bf16(v16bf a, v16bf b, v8f c) {
  return __builtin_amdgcn_wmma_f32_16x16x32_bf16(false, a, false, b, (short)0, c, false, false);
}

__device__ __forceinline__ void pack4(__bf16* u, float4 v) {
  u[0] = (__bf16)v.x; u[1] = (__bf16)v.y; u[2] = (__bf16)v.z; u[3] = (__bf16)v.w;
}

// ---------------------------------------------------------------------------
// K0: pack an f32 matrix B[k][n] (k,n in [0,128)) into per-lane WMMA B-fragment
// layout, bf16.  B element = src[k*k_stride + n*n_stride].
// Fragment layout mirrors the documented 16-bit A layout:
//   lane L: N = ntile*16 + (L&15); elems 0..7 -> K = kc*32 + (L>>4)*8 + e
//                                  elems 8..15 -> K = kc*32 + (L>>4)*8 + 16 + (e-8)
// ---------------------------------------------------------------------------
__global__ __launch_bounds__(256) void pack_b_kernel(
    const float* __restrict__ src, __bf16* __restrict__ dst,
    int k_stride, int n_stride) {
  int idx = blockIdx.x * 256 + threadIdx.x;
  if (idx >= PKELEMS) return;
  int e    = idx & 15;
  int lane = (idx >> 4) & 31;
  int kc   = (idx >> 9) & 3;
  int nt   = idx >> 11;
  int n = nt * 16 + (lane & 15);
  int k = kc * 32 + ((lane >> 4) * 8) + (e < 8 ? e : e + 8);
  dst[idx] = (__bf16)src[k * k_stride + n * n_stride];
}

// ---------------------------------------------------------------------------
// K1: per-edge transforms.
//   x_kj[e] = silu(x@w_from + b_from) * (radial@w_rbf)   (stored bf16)
//   h[e]    = silu(x@w_to   + b_to)                      (stored f32, agg base)
// Block = 16 edges x 128 cols; 8 waves, each owns one 16-col tile.
// ---------------------------------------------------------------------------
__global__ __launch_bounds__(256) void k1_edge(
    const float* __restrict__ x, const float* __restrict__ radial,
    const float* __restrict__ w_rbf,
    const float* __restrict__ b_from, const float* __restrict__ b_to,
    const __bf16* __restrict__ p_from, const __bf16* __restrict__ p_to,
    __bf16* __restrict__ x_kj, float* __restrict__ h, int E) {
  __shared__ __align__(16) float rad[16][NRAD];
  int e0 = blockIdx.x * 16;
  int t  = threadIdx.x;
  if (t < 16 * NRAD) {
    int m = t / NRAD, q = t % NRAD;
    int e = e0 + m; if (e >= E) e = E - 1;
    rad[m][q] = radial[(size_t)e * NRAD + q];
  }
  __syncthreads();
  int lane = t & 31, wave = t >> 5;
  int mrow = lane & 15;
  int kb   = (lane >> 4) * 8;      // K half-offset for A; also M base for C rows
  int n    = wave * 16 + mrow;

  int eA = e0 + mrow; if (eA >= E) eA = E - 1;
  const float* xr = x + (size_t)eA * H;
  ABf16 afr[4];
#pragma unroll
  for (int kc = 0; kc < 4; ++kc) {
    const float4* p0 = (const float4*)(xr + kc * 32 + kb);
    const float4* p1 = (const float4*)(xr + kc * 32 + kb + 16);
    pack4(afr[kc].b + 0,  p0[0]);
    pack4(afr[kc].b + 4,  p0[1]);
    pack4(afr[kc].b + 8,  p1[0]);
    pack4(afr[kc].b + 12, p1[1]);
  }

  float bfv = b_from[n], btv = b_to[n];
  v8f cf, ct;
#pragma unroll
  for (int r = 0; r < 8; ++r) { cf[r] = bfv; ct[r] = btv; }
#pragma unroll
  for (int kc = 0; kc < 4; ++kc) {
    const v16bf* bp0 = (const v16bf*)(p_from + (size_t)(((wave * 4 + kc) * 32) + lane) * 16);
    const v16bf* bp1 = (const v16bf*)(p_to   + (size_t)(((wave * 4 + kc) * 32) + lane) * 16);
    cf = wmma_bf16(afr[kc].v, *bp0, cf);
    ct = wmma_bf16(afr[kc].v, *bp1, ct);
  }

  float wr[NRAD];
#pragma unroll
  for (int q = 0; q < NRAD; ++q) wr[q] = w_rbf[q * H + n];
#pragma unroll
  for (int r = 0; r < 8; ++r) {
    int m = kb + r;
    int e = e0 + m;
    float rb = 0.f;
#pragma unroll
    for (int q = 0; q < NRAD; ++q) rb += rad[m][q] * wr[q];
    if (e < E) {
      x_kj[(size_t)e * H + n] = (__bf16)(silu(cf[r]) * rb);
      h   [(size_t)e * H + n] = silu(ct[r]);
    }
  }
}

// ---------------------------------------------------------------------------
// K2: bilinear einsum + scatter, M-blocked 4x (64 triplets/block).
//   Loop kc -> j -> mt: one B fragment in registers is replayed against
//   4 A tiles, cutting B (L2) traffic 4x.  sbf scaling is folded into the
//   A operand (row-per-lane => lane-uniform v_pk_mul_bf16 splat multiply),
//   so one f32 accumulator per M-tile suffices.
// ---------------------------------------------------------------------------
__global__ __launch_bounds__(256) void k2_bilinear(
    const __bf16* __restrict__ x_kj,
    const float* __restrict__ spherical, const float* __restrict__ w_sbf,
    const int* __restrict__ idx_from, const int* __restrict__ idx_to,
    const __bf16* __restrict__ p_W,   // 8 packed matrices, contiguous
    float* __restrict__ h, int T, int E) {
  __shared__ int from_s[64], to_s[64];
  __shared__ __align__(16) float sbf_s[64][NB];
  int t0 = blockIdx.x * 64;
  int t  = threadIdx.x;
  if (t < 64) {
    int w = t0 + t; if (w >= T) w = T - 1;
    int ef = idx_from[w]; if (ef < 0) ef = 0; if (ef >= E) ef = E - 1;
    int et = idx_to[w];   if (et < 0) et = 0; if (et >= E) et = E - 1;
    from_s[t] = ef; to_s[t] = et;
  }
  // 64 rows x 8 j = 512 dot-42 products over 256 threads (2 each)
#pragma unroll
  for (int p = 0; p < 2; ++p) {
    int idx = t + p * 256;
    int w = idx >> 3, j = idx & 7;
    int wg = t0 + w; if (wg >= T) wg = T - 1;
    const float* sp = spherical + (size_t)wg * NSNR;
    float s = 0.f;
#pragma unroll
    for (int q = 0; q < NSNR; ++q) s += sp[q] * w_sbf[q * NB + j];
    sbf_s[w][j] = s;
  }
  __syncthreads();

  int lane = t & 31, wave = t >> 5;
  int mrow = lane & 15;
  int kb   = (lane >> 4) * 8;
  int n    = wave * 16 + mrow;

  // per-lane sbf scales (A row = lane&15 in both lane halves)
  __bf16 sb[4][NB];
#pragma unroll
  for (int mt = 0; mt < 4; ++mt)
#pragma unroll
    for (int j = 0; j < NB; ++j) sb[mt][j] = (__bf16)sbf_s[mt * 16 + mrow][j];

  const __bf16* arow[4];
#pragma unroll
  for (int mt = 0; mt < 4; ++mt)
    arow[mt] = x_kj + (size_t)from_s[mt * 16 + mrow] * H;

  v8f acc[4];
#pragma unroll
  for (int mt = 0; mt < 4; ++mt)
#pragma unroll
    for (int r = 0; r < 8; ++r) acc[mt][r] = 0.f;

#pragma unroll
  for (int kc = 0; kc < 4; ++kc) {
    ABf16 af[4];
#pragma unroll
    for (int mt = 0; mt < 4; ++mt) {
      af[mt].q[0] = *(const uint4*)(arow[mt] + kc * 32 + kb);
      af[mt].q[1] = *(const uint4*)(arow[mt] + kc * 32 + kb + 16);
    }
#pragma unroll
    for (int j = 0; j < NB; ++j) {
      ABf16 bfr;
      const __bf16* bp = p_W + (size_t)j * PKELEMS
                       + (size_t)(((wave * 4 + kc) * 32) + lane) * 16;
      bfr.q[0] = *(const uint4*)bp;
      bfr.q[1] = *(const uint4*)(bp + 8);
#pragma unroll
      for (int mt = 0; mt < 4; ++mt) {
        v16bf sv;
#pragma unroll
        for (int e = 0; e < 16; ++e) sv[e] = sb[mt][j];
        v16bf as = af[mt].v * sv;          // v_pk_mul_bf16 x8
        acc[mt] = wmma_bf16(as, bfr.v, acc[mt]);
      }
    }
  }

#pragma unroll
  for (int mt = 0; mt < 4; ++mt)
#pragma unroll
    for (int r = 0; r < 8; ++r) {
      int m = mt * 16 + kb + r;
      int w = t0 + m;
      if (w < T) atomicAdd(&h[(size_t)to_s[m] * H + n], acc[mt][r]);
    }
}

// ---------------------------------------------------------------------------
// K3: residual / lin chain on h (7 chained 128x128 GEMMs per 16-row tile,
// bf16 tiles double-buffered in LDS between stages).
// ---------------------------------------------------------------------------
__device__ __forceinline__ v16bf lds_afrag(const __bf16* buf, int mrow, int kb, int kc) {
  ABf16 u;
  const __bf16* p = buf + mrow * LSTR + kc * 32 + kb;
  u.q[0] = *(const uint4*)p;
  u.q[1] = *(const uint4*)(p + 16);
  return u.v;
}

__device__ __forceinline__ v8f gemm_stage(const __bf16* buf,
    const __bf16* __restrict__ wp, const float* __restrict__ bias,
    int lane, int wave) {
  int mrow = lane & 15, kb = (lane >> 4) * 8;
  float bv = bias[wave * 16 + mrow];
  v8f c;
#pragma unroll
  for (int r = 0; r < 8; ++r) c[r] = bv;
#pragma unroll
  for (int kc = 0; kc < 4; ++kc) {
    v16bf a = lds_afrag(buf, mrow, kb, kc);
    const v16bf* bp = (const v16bf*)(wp + (size_t)(((wave * 4 + kc) * 32) + lane) * 16);
    c = wmma_bf16(a, *bp, c);
  }
  return c;
}

__device__ __forceinline__ void store_slice(__bf16* buf, v8f v, int lane, int wave) {
  int n  = wave * 16 + (lane & 15);
  int mb = (lane >> 4) * 8;
#pragma unroll
  for (int r = 0; r < 8; ++r) buf[(mb + r) * LSTR + n] = (__bf16)v[r];
}

__global__ __launch_bounds__(256) void k3_chain(
    const float* __restrict__ x, const float* __restrict__ h,
    const __bf16* __restrict__ p_rb1, const __bf16* __restrict__ p_rb2,
    const __bf16* __restrict__ p_lin,
    const __bf16* __restrict__ p_ra01, const __bf16* __restrict__ p_ra02,
    const __bf16* __restrict__ p_ra11, const __bf16* __restrict__ p_ra12,
    const float* __restrict__ rb_b, const float* __restrict__ lin_b,
    const float* __restrict__ ra_b,
    float* __restrict__ out, int E) {
  __shared__ __align__(16) __bf16 bufA[16 * LSTR];
  __shared__ __align__(16) __bf16 bufB[16 * LSTR];
  int e0 = blockIdx.x * 16;
  int t  = threadIdx.x;
  int lane = t & 31, wave = t >> 5;
  int kb = (lane >> 4) * 8, n = wave * 16 + (lane & 15);

  v8f hc;
#pragma unroll
  for (int r = 0; r < 8; ++r) {
    int e = e0 + kb + r; if (e >= E) e = E - 1;
    hc[r] = h[(size_t)e * H + n];
  }
  store_slice(bufA, hc, lane, wave);
  __syncthreads();

  // residual block 0
  v8f u = gemm_stage(bufA, p_rb1, rb_b, lane, wave);
#pragma unroll
  for (int r = 0; r < 8; ++r) u[r] = silu(u[r]);
  store_slice(bufB, u, lane, wave);
  __syncthreads();
  u = gemm_stage(bufB, p_rb2, rb_b + H, lane, wave);
#pragma unroll
  for (int r = 0; r < 8; ++r) hc[r] += silu(u[r]);
  store_slice(bufA, hc, lane, wave);
  __syncthreads();

  // lin + skip(x)
  u = gemm_stage(bufA, p_lin, lin_b, lane, wave);
#pragma unroll
  for (int r = 0; r < 8; ++r) {
    int e = e0 + kb + r; if (e >= E) e = E - 1;
    hc[r] = silu(u[r]) + x[(size_t)e * H + n];
  }
  store_slice(bufB, hc, lane, wave);
  __syncthreads();

  // residual block ra0
  u = gemm_stage(bufB, p_ra01, ra_b, lane, wave);
#pragma unroll
  for (int r = 0; r < 8; ++r) u[r] = silu(u[r]);
  store_slice(bufA, u, lane, wave);
  __syncthreads();
  u = gemm_stage(bufA, p_ra02, ra_b + H, lane, wave);
#pragma unroll
  for (int r = 0; r < 8; ++r) hc[r] += silu(u[r]);
  store_slice(bufB, hc, lane, wave);
  __syncthreads();

  // residual block ra1
  u = gemm_stage(bufB, p_ra11, ra_b + 2 * H, lane, wave);
#pragma unroll
  for (int r = 0; r < 8; ++r) u[r] = silu(u[r]);
  store_slice(bufA, u, lane, wave);
  __syncthreads();
  u = gemm_stage(bufA, p_ra12, ra_b + 3 * H, lane, wave);
#pragma unroll
  for (int r = 0; r < 8; ++r) {
    int e = e0 + kb + r;
    if (e < E) out[(size_t)e * H + n] = hc[r] + silu(u[r]);
  }
}

// ---------------------------------------------------------------------------
extern "C" void kernel_launch(void* const* d_in, const int* in_sizes, int n_in,
                              void* d_out, int out_size, void* d_ws, size_t ws_size,
                              hipStream_t stream) {
  const float* x        = (const float*)d_in[0];
  const float* radial   = (const float*)d_in[1];
  const float* spher    = (const float*)d_in[2];
  const int*   idx_from = (const int*)d_in[3];
  const int*   idx_to   = (const int*)d_in[4];
  const float* w_rbf    = (const float*)d_in[5];
  const float* w_sbf    = (const float*)d_in[6];
  const float* w_from   = (const float*)d_in[7];
  const float* b_from   = (const float*)d_in[8];
  const float* w_to     = (const float*)d_in[9];
  const float* b_to     = (const float*)d_in[10];
  const float* W        = (const float*)d_in[11];
  const float* rb_w     = (const float*)d_in[12];
  const float* rb_b     = (const float*)d_in[13];
  const float* lin_w    = (const float*)d_in[14];
  const float* lin_b    = (const float*)d_in[15];
  const float* ra_w     = (const float*)d_in[16];
  const float* ra_b     = (const float*)d_in[17];
  float* out = (float*)d_out;
  int E = in_sizes[0] / H;
  int T = in_sizes[3];

  char* ws = (char*)d_ws;
  float* h = (float*)ws;                       // [E,H] f32 (x_ji base + agg)
  size_t off = (size_t)E * H * sizeof(float);
  __bf16* x_kj = (__bf16*)(ws + off);          // [E,H] bf16
  off += (size_t)E * H * sizeof(__bf16);
  off = (off + 255) & ~(size_t)255;
  __bf16* packs = (__bf16*)(ws + off);
  __bf16* p_from = packs + 0 * (size_t)PKELEMS;
  __bf16* p_to   = packs + 1 * (size_t)PKELEMS;
  __bf16* p_rb1  = packs + 2 * (size_t)PKELEMS;
  __bf16* p_rb2  = packs + 3 * (size_t)PKELEMS;
  __bf16* p_lin  = packs + 4 * (size_t)PKELEMS;
  __bf16* p_ra01 = packs + 5 * (size_t)PKELEMS;
  __bf16* p_ra02 = packs + 6 * (size_t)PKELEMS;
  __bf16* p_ra11 = packs + 7 * (size_t)PKELEMS;
  __bf16* p_ra12 = packs + 8 * (size_t)PKELEMS;
  __bf16* p_W    = packs + 9 * (size_t)PKELEMS;  // 8 matrices contiguous

  dim3 pg((PKELEMS + 255) / 256), pb(256);
  pack_b_kernel<<<pg, pb, 0, stream>>>(w_from,           p_from, H, 1);
  pack_b_kernel<<<pg, pb, 0, stream>>>(w_to,             p_to,   H, 1);
  pack_b_kernel<<<pg, pb, 0, stream>>>(rb_w,             p_rb1,  H, 1);
  pack_b_kernel<<<pg, pb, 0, stream>>>(rb_w + H * H,     p_rb2,  H, 1);
  pack_b_kernel<<<pg, pb, 0, stream>>>(lin_w,            p_lin,  H, 1);
  pack_b_kernel<<<pg, pb, 0, stream>>>(ra_w,             p_ra01, H, 1);
  pack_b_kernel<<<pg, pb, 0, stream>>>(ra_w + 1 * H * H, p_ra02, H, 1);
  pack_b_kernel<<<pg, pb, 0, stream>>>(ra_w + 2 * H * H, p_ra11, H, 1);
  pack_b_kernel<<<pg, pb, 0, stream>>>(ra_w + 3 * H * H, p_ra12, H, 1);
  // Wt_j[k=l][n=i] = W[i*NB*H + j*H + l]  ->  src = W + j*H, k_stride=1, n_stride=NB*H
  for (int j = 0; j < NB; ++j)
    pack_b_kernel<<<pg, pb, 0, stream>>>(W + (size_t)j * H,
                                         p_W + (size_t)j * PKELEMS, 1, NB * H);

  int gbE = (E + 15) / 16;
  int gbT = (T + 63) / 64;
  k1_edge<<<dim3(gbE), dim3(256), 0, stream>>>(x, radial, w_rbf, b_from, b_to,
                                               p_from, p_to, x_kj, h, E);
  k2_bilinear<<<dim3(gbT), dim3(256), 0, stream>>>(x_kj, spher, w_sbf,
                                                   idx_from, idx_to, p_W, h, T, E);
  k3_chain<<<dim3(gbE), dim3(256), 0, stream>>>(x, h, p_rb1, p_rb2, p_lin,
                                                p_ra01, p_ra02, p_ra11, p_ra12,
                                                rb_b, lin_b, ra_b, out, E);
}
